// GAT1_29411936043250
// MI455X (gfx1250) — matrix operand order repaired and checked by reading, not compiled, not run on hardware
//
#include <hip/hip_runtime.h>
#include <hip/hip_fp16.h>

typedef __attribute__((ext_vector_type(16))) _Float16 v16h;
typedef __attribute__((ext_vector_type(8)))  _Float16 v8h;
typedef __attribute__((ext_vector_type(8)))  float    v8f;

#define N_NODES 4096
#define NFEAT   512
#define NHID    64
#define NCLASS  16
#define NHEADS  8
#define ALPHA   0.2f

// ---------------------------------------------------------------------------
// CDNA5 async global->LDS copy (ASYNCcnt path), inline asm for portability.
// LDS byte address = low 32 bits of the generic pointer (ISA 10.2 aperture map).
// ---------------------------------------------------------------------------
__device__ __forceinline__ void async_ld_b64(void* lds_ptr, const void* gptr) {
  unsigned l = (unsigned)(unsigned long long)lds_ptr;
  unsigned long long g = (unsigned long long)gptr;
  asm volatile("global_load_async_to_lds_b64 %0, %1, off" :: "v"(l), "v"(g) : "memory");
}
__device__ __forceinline__ void wait_async_le1() {
  asm volatile("s_wait_asynccnt 0x1" ::: "memory");
}
__device__ __forceinline__ void wait_async_0() {
  asm volatile("s_wait_asynccnt 0x0" ::: "memory");
}

// ---------------------------------------------------------------------------
// GEMM1: h1T[h][d][node] = (f16) sum_f x[node][f] * W1[h][f][d]
// grid (256 row-tiles, 8 heads), 128 threads = 4 waves, wave w -> d block w.
// ---------------------------------------------------------------------------
__global__ __launch_bounds__(128) void gemm1_kernel(const float* __restrict__ x,
                                                    const float* __restrict__ W1,
                                                    _Float16* __restrict__ h1T) {
  const int i0   = blockIdx.x * 16;
  const int h    = blockIdx.y;
  const int tid  = threadIdx.x;
  const int w    = tid >> 5;
  const int lane = tid & 31;
  const int m    = lane & 15;
  const int hiA  = (lane & 16) ? 8 : 0;
  const int kb   = (lane & 16) ? 16 : 0;

  __shared__ _Float16 xs[16][34];

  v8f acc; for (int i = 0; i < 8; ++i) acc[i] = 0.f;

  for (int k0 = 0; k0 < NFEAT; k0 += 32) {
    {
      const int row = tid >> 3;
      const int c4  = (tid & 7) * 4;
      const float4 v = *(const float4*)(x + (size_t)(i0 + row) * NFEAT + k0 + c4);
      xs[row][c4 + 0] = (_Float16)v.x;
      xs[row][c4 + 1] = (_Float16)v.y;
      xs[row][c4 + 2] = (_Float16)v.z;
      xs[row][c4 + 3] = (_Float16)v.w;
    }
    __syncthreads();

    v16h a, b;
    for (int i = 0; i < 16; ++i) {
      const int kk = kb + i;
      a[i] = xs[m][kk];
    }
    for (int i = 0; i < 16; ++i)
      b[i] = (_Float16)W1[((size_t)h * NFEAT + k0 + kb + i) * NHID + (w * 16 + m)];
    acc = __builtin_amdgcn_wmma_f32_16x16x32_f16(false, a, false, b, (short)0, acc,
                                                 false, false);
    __syncthreads();
  }

  const int d        = w * 16 + m;
  const int nodebase = i0 + ((lane >> 4) & 1) * 8;
  v8h st;
  for (int r = 0; r < 8; ++r) st[r] = (_Float16)acc[r];
  *(v8h*)(h1T + ((size_t)h * NHID + d) * N_NODES + nodebase) = st;
}

// ---------------------------------------------------------------------------
// f_src[h][n] = sum_d hT[h][d][n]*a_src[h][d]  (same for dst)
// ---------------------------------------------------------------------------
__global__ void fvec_kernel(const _Float16* __restrict__ hT,
                            const float* __restrict__ a_src,
                            const float* __restrict__ a_dst,
                            float* __restrict__ f_src, float* __restrict__ f_dst,
                            int D) {
  const int gid = blockIdx.x * blockDim.x + threadIdx.x;
  if (gid >= NHEADS * N_NODES) return;
  const int h = gid >> 12, n = gid & (N_NODES - 1);
  float s = 0.f, t = 0.f;
  for (int d = 0; d < D; ++d) {
    const float hv = (float)hT[((size_t)h * D + d) * N_NODES + n];
    s += hv * a_src[h * D + d];
    t += hv * a_dst[h * D + d];
  }
  f_src[gid] = s;
  f_dst[gid] = t;
}

__global__ void maxfd_kernel(const float* __restrict__ f_dst, float* __restrict__ mx) {
  __shared__ float sm[256];
  const int h = blockIdx.x;
  float m = -3.0e38f;
  for (int j = threadIdx.x; j < N_NODES; j += 256)
    m = fmaxf(m, f_dst[h * N_NODES + j]);
  sm[threadIdx.x] = m;
  __syncthreads();
  for (int s = 128; s > 0; s >>= 1) {
    if (threadIdx.x < s) sm[threadIdx.x] = fmaxf(sm[threadIdx.x], sm[threadIdx.x + s]);
    __syncthreads();
  }
  if (threadIdx.x == 0) mx[h] = sm[0];
}

// ---------------------------------------------------------------------------
// AGG1: out1[node][h*64+d] = relu( softmax_j(masked lrelu(fs+fd)) @ h1 )  (f16)
// 8 waves = 8 heads share a double-buffered async-staged adjacency tile.
// ---------------------------------------------------------------------------
__global__ __launch_bounds__(256) void agg1_kernel(const float* __restrict__ adj,
                                                   const _Float16* __restrict__ h1T,
                                                   const float* __restrict__ f_src,
                                                   const float* __restrict__ f_dst,
                                                   const float* __restrict__ mxfd,
                                                   const int* __restrict__ thr,
                                                   _Float16* __restrict__ out1) {
  const int i0   = blockIdx.x * 16;
  const int tid  = threadIdx.x;
  const int h    = tid >> 5;
  const int lane = tid & 31;
  const int m    = lane & 15;
  const int hiA  = (lane & 16) ? 8 : 0;
  const int kb   = (lane & 16) ? 16 : 0;
  const float thresh = (float)(*thr);

  __shared__ float adjt[2][16][34];       // even stride -> aligned b64 pairs
  __shared__ float fdsh[NHEADS][32];
  __shared__ float rssh[NHEADS][16];

  const int srow = tid >> 4;              // staging: row 0..15
  const int scol = (tid & 15) * 2;        // staging: col pair

  const float fsrow = f_src[h * N_NODES + i0 + m];
  float moff = fsrow + mxfd[h];
  moff = moff > 0.f ? moff : ALPHA * moff;   // >= lrelu(fs+fd) for all j

  v8f acc[4];
  for (int b2 = 0; b2 < 4; ++b2)
    for (int i = 0; i < 8; ++i) acc[b2][i] = 0.f;
  float rowpart = 0.f;

  // prime buffer 0
  async_ld_b64(&adjt[0][srow][scol],
               adj + (size_t)(i0 + srow) * N_NODES + 0 + scol);

  for (int j0 = 0; j0 < N_NODES; j0 += 32) {
    const int cur = (j0 >> 5) & 1;
    const bool more = (j0 + 32 < N_NODES);
    if (more)
      async_ld_b64(&adjt[cur ^ 1][srow][scol],
                   adj + (size_t)(i0 + srow) * N_NODES + (j0 + 32) + scol);
    if (more) wait_async_le1(); else wait_async_0();
    fdsh[h][lane] = f_dst[h * N_NODES + j0 + lane];  // per-head fd slice (own wave)
    __syncthreads();

    v16h a;
    for (int p = 0; p < 8; ++p) {
      const int kk0 = kb + 2 * p;
      const int ia  = 2 * p;
      const float2 av = *(const float2*)&adjt[cur][m][kk0];
      const float2 fv = *(const float2*)&fdsh[h][kk0];
      float t0 = fsrow + fv.x, t1 = fsrow + fv.y;
      t0 = t0 > 0.f ? t0 : ALPHA * t0;
      t1 = t1 > 0.f ? t1 : ALPHA * t1;
      const float p0 = (av.x > thresh) ? __expf(t0 - moff) : 0.f;
      const float p1 = (av.y > thresh) ? __expf(t1 - moff) : 0.f;
      rowpart += p0 + p1;
      a[ia]     = (_Float16)p0;
      a[ia + 1] = (_Float16)p1;
    }
    for (int b2 = 0; b2 < 4; ++b2) {
      const v16h bfrag =
          *(const v16h*)(h1T + ((size_t)h * NHID + b2 * 16 + m) * N_NODES + j0 + kb);
      acc[b2] = __builtin_amdgcn_wmma_f32_16x16x32_f16(false, a, false, bfrag, (short)0,
                                                       acc[b2], false, false);
    }
    __syncthreads();
  }

  const float tot = rowpart + __shfl_xor(rowpart, 16, 32);
  if (lane < 16) rssh[h][lane] = fmaxf(tot, 1e-30f);
  __syncthreads();

  const int rowb = ((lane >> 4) & 1) * 8;
  for (int b2 = 0; b2 < 4; ++b2)
    for (int r = 0; r < 8; ++r) {
      const int row = rowb + r;
      float v = acc[b2][r] / rssh[h][row];
      v = fmaxf(v, 0.f);
      out1[(size_t)(i0 + row) * (NHEADS * NHID) + h * NHID + b2 * 16 + m] = (_Float16)v;
    }
}

// ---------------------------------------------------------------------------
// GEMM2: h2T[h][c][node] = (f16) sum_f out1[node][f] * W2[h][f][c]; 1 wave/block
// ---------------------------------------------------------------------------
__global__ __launch_bounds__(32) void gemm2_kernel(const _Float16* __restrict__ out1,
                                                   const float* __restrict__ W2,
                                                   _Float16* __restrict__ h2T) {
  const int i0   = blockIdx.x * 16;
  const int h    = blockIdx.y;
  const int lane = threadIdx.x & 31;
  const int m    = lane & 15;
  const int hiA  = (lane & 16) ? 8 : 0;
  const int kb   = (lane & 16) ? 16 : 0;

  v8f acc; for (int i = 0; i < 8; ++i) acc[i] = 0.f;

  for (int k0 = 0; k0 < NFEAT; k0 += 32) {
    v16h a, b;
    const v8h a0 = *(const v8h*)(out1 + (size_t)(i0 + m) * NFEAT + k0 + hiA);
    const v8h a1 = *(const v8h*)(out1 + (size_t)(i0 + m) * NFEAT + k0 + 16 + hiA);
    for (int i = 0; i < 8; ++i) { a[i] = a0[i]; a[8 + i] = a1[i]; }
    for (int i = 0; i < 16; ++i)
      b[i] = (_Float16)W2[((size_t)h * NFEAT + k0 + (i < 8 ? i : 8 + i) + hiA) * NCLASS + m];
    acc = __builtin_amdgcn_wmma_f32_16x16x32_f16(false, a, false, b, (short)0, acc,
                                                 false, false);
  }
  v8h st; for (int r = 0; r < 8; ++r) st[r] = (_Float16)acc[r];
  *(v8h*)(h2T + ((size_t)h * NCLASS + m) * N_NODES + i0 + ((lane >> 4) & 1) * 8) = st;
}

// ---------------------------------------------------------------------------
// AGG2: out[node][c] = relu( mean_h softmax-aggregate(h2) ), f32 output
// ---------------------------------------------------------------------------
__global__ __launch_bounds__(256) void agg2_kernel(const float* __restrict__ adj,
                                                   const _Float16* __restrict__ h2T,
                                                   const float* __restrict__ f_src,
                                                   const float* __restrict__ f_dst,
                                                   const float* __restrict__ mxfd,
                                                   const int* __restrict__ thr,
                                                   float* __restrict__ outp) {
  const int i0   = blockIdx.x * 16;
  const int tid  = threadIdx.x;
  const int h    = tid >> 5;
  const int lane = tid & 31;
  const int m    = lane & 15;
  const int hiA  = (lane & 16) ? 8 : 0;
  const int kb   = (lane & 16) ? 16 : 0;
  const float thresh = (float)(*thr);

  __shared__ float adjt[2][16][34];
  __shared__ float fdsh[NHEADS][32];
  __shared__ float rssh[NHEADS][16];
  __shared__ float redbuf[256];

  redbuf[tid] = 0.f;

  const int srow = tid >> 4;
  const int scol = (tid & 15) * 2;

  const float fsrow = f_src[h * N_NODES + i0 + m];
  float moff = fsrow + mxfd[h];
  moff = moff > 0.f ? moff : ALPHA * moff;

  v8f acc; for (int i = 0; i < 8; ++i) acc[i] = 0.f;
  float rowpart = 0.f;

  async_ld_b64(&adjt[0][srow][scol],
               adj + (size_t)(i0 + srow) * N_NODES + 0 + scol);

  for (int j0 = 0; j0 < N_NODES; j0 += 32) {
    const int cur = (j0 >> 5) & 1;
    const bool more = (j0 + 32 < N_NODES);
    if (more)
      async_ld_b64(&adjt[cur ^ 1][srow][scol],
                   adj + (size_t)(i0 + srow) * N_NODES + (j0 + 32) + scol);
    if (more) wait_async_le1(); else wait_async_0();
    fdsh[h][lane] = f_dst[h * N_NODES + j0 + lane];
    __syncthreads();

    v16h a;
    for (int p = 0; p < 8; ++p) {
      const int kk0 = kb + 2 * p;
      const int ia  = 2 * p;
      const float2 av = *(const float2*)&adjt[cur][m][kk0];
      const float2 fv = *(const float2*)&fdsh[h][kk0];
      float t0 = fsrow + fv.x, t1 = fsrow + fv.y;
      t0 = t0 > 0.f ? t0 : ALPHA * t0;
      t1 = t1 > 0.f ? t1 : ALPHA * t1;
      const float p0 = (av.x > thresh) ? __expf(t0 - moff) : 0.f;
      const float p1 = (av.y > thresh) ? __expf(t1 - moff) : 0.f;
      rowpart += p0 + p1;
      a[ia]     = (_Float16)p0;
      a[ia + 1] = (_Float16)p1;
    }
    const v16h bfrag = *(const v16h*)(h2T + ((size_t)h * NCLASS + m) * N_NODES + j0 + kb);
    acc = __builtin_amdgcn_wmma_f32_16x16x32_f16(false, a, false, bfrag, (short)0, acc,
                                                 false, false);
    __syncthreads();
  }

  const float tot = rowpart + __shfl_xor(rowpart, 16, 32);
  if (lane < 16) rssh[h][lane] = fmaxf(tot, 1e-30f);
  __syncthreads();

  const int rowb = ((lane >> 4) & 1) * 8;
  for (int r = 0; r < 8; ++r) {
    const float v = acc[r] / rssh[h][rowb + r];
    atomicAdd(&redbuf[(rowb + r) * 16 + m], v);
  }
  __syncthreads();
  {
    const int node = i0 + (tid >> 4), cls = tid & 15;
    outp[(size_t)node * NCLASS + cls] = fmaxf(redbuf[tid] * (1.0f / NHEADS), 0.f);
  }
}

// ---------------------------------------------------------------------------
extern "C" void kernel_launch(void* const* d_in, const int* in_sizes, int n_in,
                              void* d_out, int out_size, void* d_ws, size_t ws_size,
                              hipStream_t stream) {
  const float* x   = (const float*)d_in[0];
  const float* adj = (const float*)d_in[1];
  const float* W1  = (const float*)d_in[2];
  const float* a1s = (const float*)d_in[3];
  const float* a1d = (const float*)d_in[4];
  const float* W2  = (const float*)d_in[5];
  const float* a2s = (const float*)d_in[6];
  const float* a2d = (const float*)d_in[7];
  const int*   thr = (const int*)d_in[8];
  float* out = (float*)d_out;

  char* p = (char*)d_ws;
  _Float16* h1T  = (_Float16*)p; p += (size_t)NHEADS * NHID * N_NODES * 2;
  _Float16* out1 = (_Float16*)p; p += (size_t)N_NODES * NHEADS * NHID * 2;
  _Float16* h2T  = (_Float16*)p; p += (size_t)NHEADS * NCLASS * N_NODES * 2;
  float* fs1 = (float*)p; p += (size_t)NHEADS * N_NODES * 4;
  float* fd1 = (float*)p; p += (size_t)NHEADS * N_NODES * 4;
  float* fs2 = (float*)p; p += (size_t)NHEADS * N_NODES * 4;
  float* fd2 = (float*)p; p += (size_t)NHEADS * N_NODES * 4;
  float* mx1 = (float*)p; p += 256;
  float* mx2 = (float*)p; p += 256;

  const int ntiles = N_NODES / 16;  // 256

  gemm1_kernel<<<dim3(ntiles, NHEADS), 128, 0, stream>>>(x, W1, h1T);
  fvec_kernel<<<(NHEADS * N_NODES) / 256, 256, 0, stream>>>(h1T, a1s, a1d, fs1, fd1, NHID);
  maxfd_kernel<<<NHEADS, 256, 0, stream>>>(fd1, mx1);
  agg1_kernel<<<ntiles, 256, 0, stream>>>(adj, h1T, fs1, fd1, mx1, thr, out1);

  gemm2_kernel<<<dim3(ntiles, NHEADS), 32, 0, stream>>>(out1, W2, h2T);
  fvec_kernel<<<(NHEADS * N_NODES) / 256, 256, 0, stream>>>(h2T, a2s, a2d, fs2, fd2, NCLASS);
  maxfd_kernel<<<NHEADS, 256, 0, stream>>>(fd2, mx2);
  agg2_kernel<<<ntiles, 256, 0, stream>>>(adj, h2T, fs2, fd2, mx2, thr, out);
}
